// EncoderModule_19000935317882
// MI455X (gfx1250) — matrix operand order repaired and checked by reading, not compiled, not run on hardware
//
#include <hip/hip_runtime.h>
#include <math.h>

// ---------------- problem constants ----------------
#define N_    4096
#define E_    32768
#define B_    1024
#define NF_   32
#define EF_   16
#define ENC_  64
#define HID_  128
#define HEADS_ 4
#define MAXS_ 10
#define KA_   640      // 5*HID
#define KM_   384      // 3*HID
#define NOUT_ 512      // HEADS*HID
#define ASTRIDE_ 648   // LDS A-tile row stride in halves (padded, 16B-aligned rows)

typedef __attribute__((ext_vector_type(16))) __bf16 v16bf;
typedef __attribute__((ext_vector_type(8)))  float  v8f;

union FragB { uint4 u[2]; v16bf v; };

static __device__ __forceinline__ unsigned short f2bf(float f) {
  union { float f; unsigned u; } x; x.f = f;
  unsigned r = x.u + 0x7fffu + ((x.u >> 16) & 1u);   // round-to-nearest-even
  return (unsigned short)(r >> 16);
}

static __device__ __forceinline__ float atomicMaxFloat(float* addr, float val) {
  int* a = (int*)addr;
  int old = __float_as_int(*addr);
  while (__int_as_float(old) < val) {
    int assumed = old;
    old = atomicCAS(a, assumed, __float_as_int(val));
    if (old == assumed) break;
  }
  return __int_as_float(old);
}

// ---------------- prep kernels ----------------

// h0 = nf @ W_h ; h = h0 ; node_in(bf16) = [h | h0]
__global__ __launch_bounds__(256) void k_prep_nodes(
    const float* __restrict__ nf, const float* __restrict__ W_h,
    float* __restrict__ h, unsigned short* __restrict__ nodein) {
  int idx = blockIdx.x * blockDim.x + threadIdx.x;
  if (idx >= N_ * HID_) return;
  int n = idx >> 7, c = idx & 127;
  float s = 0.f;
  #pragma unroll
  for (int k = 0; k < NF_; ++k) s += nf[n * NF_ + k] * W_h[k * HID_ + c];
  h[idx] = s;
  unsigned short b = f2bf(s);
  nodein[n * 256 + c]       = b;   // h part (updated each iter)
  nodein[n * 256 + 128 + c] = b;   // h0 part (static)
}

// e(bf16) = ef @ W_e ; mask = any(ef != 0)
__global__ __launch_bounds__(256) void k_prep_edges(
    const float* __restrict__ ef, const float* __restrict__ W_e,
    unsigned short* __restrict__ ebf, int* __restrict__ mask) {
  int idx = blockIdx.x * blockDim.x + threadIdx.x;
  if (idx >= E_ * HID_) return;
  int e = idx >> 7, c = idx & 127;
  float s = 0.f;
  #pragma unroll
  for (int k = 0; k < EF_; ++k) s += ef[e * EF_ + k] * W_e[k * HID_ + c];
  ebf[idx] = f2bf(s);
  if (c == 0) {
    int m = 0;
    #pragma unroll
    for (int k = 0; k < EF_; ++k) m |= (ef[e * EF_ + k] != 0.f) ? 1 : 0;
    mask[e] = m;
  }
}

// column-major bf16 weights: Wat[n*640+k] = W_a[k*512+n]; Wmt[n*384+k] = W_m[k*512+n]
__global__ __launch_bounds__(256) void k_prep_weights(
    const float* __restrict__ W_a, const float* __restrict__ W_m,
    unsigned short* __restrict__ Wat, unsigned short* __restrict__ Wmt) {
  int idx = blockIdx.x * blockDim.x + threadIdx.x;
  if (idx < KA_ * NOUT_) {
    int k = idx / NOUT_, n = idx % NOUT_;
    Wat[n * KA_ + k] = f2bf(W_a[idx]);
  } else if (idx < (KA_ + KM_) * NOUT_) {
    int j = idx - KA_ * NOUT_;
    int k = j / NOUT_, n = j % NOUT_;
    Wmt[n * KM_ + k] = f2bf(W_m[j]);
  }
}

// per-iteration reset
__global__ __launch_bounds__(256) void k_iter_init(
    float* __restrict__ mx, float* __restrict__ den, float* __restrict__ agg) {
  int idx = blockIdx.x * blockDim.x + threadIdx.x;
  if (idx < N_ * HID_) agg[idx] = 0.f;
  if (idx < N_ * HEADS_) { mx[idx] = -INFINITY; den[idx] = 0.f; }
}

// ---------------- main WMMA edge GEMM ----------------
// Block: 16 edges, 256 threads (8 waves).
// k-outer / tile-inner: one A fragment per k-step feeds 8 back-to-back WMMAs
// into 8 independent accumulators (no D->A/B hazards, 8x less LDS traffic).
// Waves 0-3: head = wave; 8 N-tiles of h_a (K=640); fused leaky_relu + a_kern dot.
// Waves 4-7: 8 N-tiles each of h_m (K=384) -> hm[E,512] fp32.
__global__ __launch_bounds__(256, 2) void k_edge_gemm(
    const unsigned short* __restrict__ nodein,  // [N,256] bf16
    const unsigned short* __restrict__ ebf,     // [E,128] bf16
    const int* __restrict__ eidx,               // [2,E]
    const unsigned short* __restrict__ Wat,     // [512,640] bf16 col-major-of-W_a
    const unsigned short* __restrict__ Wmt,     // [512,384] bf16
    const float* __restrict__ akern,            // [512]
    float* __restrict__ logits,                 // [E,4]
    float* __restrict__ hm)                     // [E,512]
{
  __shared__ __align__(16) unsigned short As[16 * ASTRIDE_];
  __shared__ int sIdx[32];

  const int eBase = blockIdx.x * 16;
  if (threadIdx.x < 16) {
    sIdx[threadIdx.x]      = eidx[eBase + threadIdx.x];        // src
    sIdx[16 + threadIdx.x] = eidx[E_ + eBase + threadIdx.x];   // tgt
  }
  __syncthreads();

  // stage A rows: [ node_in[src](256) | e(128) | node_in[tgt](256) ] in bf16
  for (int i = threadIdx.x; i < 16 * 80; i += 256) {
    int row = i / 80;
    int ho  = (i % 80) * 8;
    uint4 v;
    if (ho < 256)      v = *(const uint4*)(nodein + sIdx[row] * 256 + ho);
    else if (ho < 384) v = *(const uint4*)(ebf + (eBase + row) * 128 + (ho - 256));
    else               v = *(const uint4*)(nodein + sIdx[16 + row] * 256 + (ho - 384));
    *(uint4*)(As + row * ASTRIDE_ + ho) = v;
  }
  __syncthreads();

  const int wave   = threadIdx.x >> 5;
  const int lane   = threadIdx.x & 31;
  const int halfId = lane >> 4;     // A/B: K-run base 0/8
  const int lan    = lane & 15;     // A: row M ; B/C/D: column N

  if (wave < 4) {
    // ---- h_a : one head per wave, 8 column tiles, K = 640 ----
    v8f acc[8] = {};
    const unsigned short* arow  = As + lan * ASTRIDE_ + halfId * 8;
    const unsigned short* bbase = Wat + (wave * 128 + lan) * KA_ + halfId * 8;
    for (int kk = 0; kk < KA_; kk += 32) {
      FragB a;
      a.u[0] = *(const uint4*)(arow + kk);
      a.u[1] = *(const uint4*)(arow + kk + 16);
      #pragma unroll
      for (int t = 0; t < 8; ++t) {
        FragB b;
        const unsigned short* bp = bbase + t * (16 * KA_) + kk;
        b.u[0] = *(const uint4*)(bp);
        b.u[1] = *(const uint4*)(bp + 16);
        acc[t] = __builtin_amdgcn_wmma_f32_16x16x32_bf16(
            false, a.v, false, b.v, (short)0, acc[t], false, false);
      }
    }
    // fused leaky_relu(0.2) + dot with a_kern, accumulated over this wave's 128 cols
    float pacc[8] = {0.f, 0.f, 0.f, 0.f, 0.f, 0.f, 0.f, 0.f};
    #pragma unroll
    for (int t = 0; t < 8; ++t) {
      const float ak = akern[(wave * 8 + t) * 16 + lan];
      #pragma unroll
      for (int r = 0; r < 8; ++r) {
        float v = acc[t][r];
        v = v > 0.f ? v : 0.2f * v;
        pacc[r] += v * ak;
      }
    }
    // reduce 16 columns-per-lane-group -> per-edge logit (halves independent)
    #pragma unroll
    for (int off = 1; off < 16; off <<= 1)
      #pragma unroll
      for (int r = 0; r < 8; ++r) pacc[r] += __shfl_xor(pacc[r], off, 32);
    if (lan == 0) {
      const int M0 = halfId * 8;
      #pragma unroll
      for (int r = 0; r < 8; ++r)
        logits[(eBase + M0 + r) * HEADS_ + wave] = pacc[r];
    }
  } else {
    // ---- h_m : 8 column tiles per wave, K = 384, A cols start at 256 ----
    v8f acc[8] = {};
    const unsigned short* arow  = As + lan * ASTRIDE_ + 256 + halfId * 8;
    const unsigned short* bbase = Wmt + ((wave - 4) * 128 + lan) * KM_ + halfId * 8;
    for (int kk = 0; kk < KM_; kk += 32) {
      FragB a;
      a.u[0] = *(const uint4*)(arow + kk);
      a.u[1] = *(const uint4*)(arow + kk + 16);
      #pragma unroll
      for (int t = 0; t < 8; ++t) {
        FragB b;
        const unsigned short* bp = bbase + t * (16 * KM_) + kk;
        b.u[0] = *(const uint4*)(bp);
        b.u[1] = *(const uint4*)(bp + 16);
        acc[t] = __builtin_amdgcn_wmma_f32_16x16x32_bf16(
            false, a.v, false, b.v, (short)0, acc[t], false, false);
      }
    }
    const int M0 = halfId * 8;
    #pragma unroll
    for (int t = 0; t < 8; ++t) {
      const int col = ((wave - 4) * 8 + t) * 16 + lan;
      #pragma unroll
      for (int r = 0; r < 8; ++r)
        hm[(eBase + M0 + r) * NOUT_ + col] = acc[t][r];  // lanes 0-15 contiguous
    }
  }
}

// ---------------- segment softmax ----------------
__global__ __launch_bounds__(256) void k_seg_max(
    const float* __restrict__ logits, const int* __restrict__ mask,
    const int* __restrict__ eidx, float* __restrict__ mx) {
  int idx = blockIdx.x * blockDim.x + threadIdx.x;
  if (idx >= E_ * HEADS_) return;
  int e = idx >> 2;
  float l = mask[e] ? logits[idx] : -INFINITY;
  atomicMaxFloat(&mx[eidx[e] * HEADS_ + (idx & 3)], l);
}

__global__ __launch_bounds__(256) void k_seg_exp(
    const float* __restrict__ logits, const int* __restrict__ mask,
    const int* __restrict__ eidx, const float* __restrict__ mx,
    float* __restrict__ ex, float* __restrict__ den) {
  int idx = blockIdx.x * blockDim.x + threadIdx.x;
  if (idx >= E_ * HEADS_) return;
  int e = idx >> 2, hh = idx & 3;
  int s = eidx[e];
  float m0 = mx[s * HEADS_];
  bool dead = isinf(m0) && (m0 < 0.f);                   // all edges of src masked
  float l = mask[e] ? logits[idx] : -INFINITY;
  float safe = dead ? 0.f : l;
  float m = dead ? 0.f : mx[s * HEADS_ + hh];
  float v = expf(safe - m);                              // exp(-inf) = 0
  ex[idx] = v;
  atomicAdd(&den[s * HEADS_ + hh], v);
}

__global__ __launch_bounds__(256) void k_aggregate(
    const float* __restrict__ ex, const float* __restrict__ den,
    const float* __restrict__ hm, const int* __restrict__ eidx,
    float* __restrict__ agg) {
  int idx = blockIdx.x * blockDim.x + threadIdx.x;
  if (idx >= E_ * HID_) return;
  int e = idx >> 7, c = idx & 127;
  int s = eidx[e];
  float acc = 0.f;
  #pragma unroll
  for (int hh = 0; hh < HEADS_; ++hh) {
    float a = ex[e * HEADS_ + hh] / den[s * HEADS_ + hh];
    acc += a * hm[e * NOUT_ + hh * HID_ + c];
  }
  atomicAdd(&agg[s * HID_ + c], acc);
}

__global__ __launch_bounds__(256) void k_node_update(
    const float* __restrict__ agg, const int* __restrict__ nst,
    float* __restrict__ h, unsigned short* __restrict__ nodein, int iter) {
  int idx = blockIdx.x * blockDim.x + threadIdx.x;
  if (idx >= N_ * HID_) return;
  int n = idx >> 7, c = idx & 127;
  float v = tanhf(agg[idx]);
  float hv = (iter < nst[n]) ? v : h[idx];
  h[idx] = hv;
  nodein[n * 256 + c] = f2bf(hv);
}

// ---------------- final projection ----------------
__global__ __launch_bounds__(256) void k_final(
    const float* __restrict__ h, const int* __restrict__ cs,
    const float* __restrict__ W_g, float* __restrict__ out) {
  int idx = blockIdx.x * blockDim.x + threadIdx.x;
  if (idx >= B_ * ENC_) return;
  int b = idx >> 6, c = idx & 63;
  const float* hr = h + cs[b] * HID_;
  float s = 0.f;
  #pragma unroll 8
  for (int k = 0; k < HID_; ++k) s += hr[k] * W_g[k * ENC_ + c];
  out[idx] = s;
}

// ---------------- launcher ----------------
extern "C" void kernel_launch(void* const* d_in, const int* in_sizes, int n_in,
                              void* d_out, int out_size, void* d_ws, size_t ws_size,
                              hipStream_t stream) {
  (void)in_sizes; (void)n_in; (void)out_size; (void)ws_size;

  const float* nf    = (const float*)d_in[0];
  const float* ef    = (const float*)d_in[1];
  const int*   eidx  = (const int*)d_in[2];
  const int*   cs    = (const int*)d_in[3];
  const int*   nst   = (const int*)d_in[4];
  const float* W_h   = (const float*)d_in[5];
  const float* W_e   = (const float*)d_in[6];
  const float* W_a   = (const float*)d_in[7];
  const float* W_m   = (const float*)d_in[8];
  const float* akern = (const float*)d_in[9];
  const float* W_g   = (const float*)d_in[10];
  float* out = (float*)d_out;

  char* w = (char*)d_ws;
  float*          d_h      = (float*)(w);                        //  2,097,152 B
  unsigned short* d_nodein = (unsigned short*)(w +  2097152);    //  2,097,152
  unsigned short* d_e      = (unsigned short*)(w +  4194304);    //  8,388,608
  int*            d_mask   = (int*)(w + 12582912);               //    131,072
  unsigned short* d_Wat    = (unsigned short*)(w + 12713984);    //    655,360
  unsigned short* d_Wmt    = (unsigned short*)(w + 13369344);    //    393,216
  float*          d_logits = (float*)(w + 13762560);             //    524,288
  float*          d_ex     = (float*)(w + 14286848);             //    524,288
  float*          d_max    = (float*)(w + 14811136);             //     65,536
  float*          d_den    = (float*)(w + 14876672);             //     65,536
  float*          d_agg    = (float*)(w + 14942208);             //  2,097,152
  float*          d_hm     = (float*)(w + 17039360);             // 67,108,864
  // total workspace: 84,148,224 bytes

  dim3 blk(256);
  k_prep_nodes<<<(N_ * HID_) / 256, blk, 0, stream>>>(nf, W_h, d_h, d_nodein);
  k_prep_edges<<<(E_ * HID_) / 256, blk, 0, stream>>>(ef, W_e, d_e, d_mask);
  k_prep_weights<<<((KA_ + KM_) * NOUT_) / 256, blk, 0, stream>>>(W_a, W_m, d_Wat, d_Wmt);

  for (int it = 0; it < MAXS_; ++it) {
    k_iter_init<<<(N_ * HID_) / 256, blk, 0, stream>>>(d_max, d_den, d_agg);
    k_edge_gemm<<<E_ / 16, blk, 0, stream>>>(d_nodein, d_e, eidx, d_Wat, d_Wmt,
                                             akern, d_logits, d_hm);
    k_seg_max<<<(E_ * HEADS_) / 256, blk, 0, stream>>>(d_logits, d_mask, eidx, d_max);
    k_seg_exp<<<(E_ * HEADS_) / 256, blk, 0, stream>>>(d_logits, d_mask, eidx, d_max,
                                                       d_ex, d_den);
    k_aggregate<<<(E_ * HID_) / 256, blk, 0, stream>>>(d_ex, d_den, d_hm, eidx, d_agg);
    k_node_update<<<(N_ * HID_) / 256, blk, 0, stream>>>(d_agg, nst, d_h, d_nodein, it);
  }

  k_final<<<(B_ * ENC_) / 256, blk, 0, stream>>>(d_h, cs, W_g, out);
}
